// PEER_11458972745884
// MI455X (gfx1250) — matrix operand order repaired and checked by reading, not compiled, not run on hardware
//
#include <hip/hip_runtime.h>
#include <hip/hip_bf16.h>

// ---------------------------------------------------------------------------
// PEER product-key expert layer for MI455X (gfx1250, wave32, WMMA).
// B=2, S=256, DIN=DOUT=1024, H=8, DQ=256, K=16, PK0=PK1=256, NE=65536, E=1
// ---------------------------------------------------------------------------

#define BQ_B 2
#define BQ_S 256
#define DIN 1024
#define DOUT 1024
#define NH 8
#define DQ 256
#define TOPK 16
#define PK 256
#define NROWS (BQ_B * BQ_S * NH)   // 4096 score rows
#define NTOK  (BQ_B * BQ_S)        // 512 tokens
#define HDQ   (NH * DQ)            // 2048

typedef __attribute__((ext_vector_type(16))) __bf16 v16bf;
typedef __attribute__((ext_vector_type(8)))  float  v8f;
typedef __attribute__((ext_vector_type(4)))  float  f4v;

union FragU {
    v16bf v;
    uint4 q[2];
};

__device__ __forceinline__ unsigned short f2bf(float x) {
    unsigned int u = __float_as_uint(x);
    unsigned int r = u + 0x7FFFu + ((u >> 16) & 1u);   // round-to-nearest-even
    return (unsigned short)(r >> 16);
}

// ---------------------------------------------------------------------------
// fp32 -> bf16 bulk convert
// ---------------------------------------------------------------------------
__global__ void k_cvt_bf16(const float* __restrict__ src,
                           unsigned short* __restrict__ dst, int n) {
    int i = blockIdx.x * blockDim.x + threadIdx.x;
    if (i < n) dst[i] = f2bf(src[i]);
}

// ---------------------------------------------------------------------------
// WMMA bf16 GEMM:  C[M,N] = A[M,Kd] * Bm[N,Kd]^T (+ bias[N])
// One wave -> 16x64 tile (4 accumulators, A fragment reused 4x).
// ---------------------------------------------------------------------------
__device__ __forceinline__ v16bf load_frag_A(const unsigned short* __restrict__ A,
                                             int lda, int row, int k0, int kh) {
    FragU f;
    const unsigned short* p = A + (size_t)row * lda + k0 + kh * 8;
    f.q[0] = *(const uint4*)(p);        // K = k0+kh*8     .. +7
    f.q[1] = *(const uint4*)(p + 16);   // K = k0+kh*8+16  .. +7
    return f.v;
}

__device__ __forceinline__ v16bf load_frag_B(const unsigned short* __restrict__ Bm,
                                             int ldb, int col, int k0, int kh) {
    FragU f;
    const unsigned short* p = Bm + (size_t)col * ldb + k0 + kh * 8;
    f.q[0] = *(const uint4*)(p);        // K = k0+kh*8     .. +7
    f.q[1] = *(const uint4*)(p + 16);   // K = k0+kh*8+16  .. +7
    return f.v;
}

__global__ void k_wmma_gemm_bf16(const unsigned short* __restrict__ A,
                                 const unsigned short* __restrict__ Bm,
                                 float* __restrict__ C,
                                 const float* __restrict__ bias,
                                 int M, int N, int Kd) {
    const int lane = threadIdx.x & 31;
    const int wave = threadIdx.x >> 5;
    const int gw = blockIdx.x * (blockDim.x >> 5) + wave;
    const int nSupers = N >> 6;                // 64-wide N strips
    const int mt = gw / nSupers;
    const int nS = gw % nSupers;
    if (mt * 16 >= M) return;

    const int kh   = lane >> 4;
    const int l15  = lane & 15;
    const int mrow = mt * 16 + l15;
    const int cb   = nS * 64;                  // column base of this strip

    v8f acc0 = {}, acc1 = {}, acc2 = {}, acc3 = {};

    for (int k0 = 0; k0 < Kd; k0 += 32) {
        v16bf a  = load_frag_A(A, Kd, mrow, k0, kh);
        v16bf b0 = load_frag_B(Bm, Kd, cb +      l15, k0, kh);
        v16bf b1 = load_frag_B(Bm, Kd, cb + 16 + l15, k0, kh);
        v16bf b2 = load_frag_B(Bm, Kd, cb + 32 + l15, k0, kh);
        v16bf b3 = load_frag_B(Bm, Kd, cb + 48 + l15, k0, kh);
        acc0 = __builtin_amdgcn_wmma_f32_16x16x32_bf16(false, a, false, b0,
                                                       (short)0, acc0, false, false);
        acc1 = __builtin_amdgcn_wmma_f32_16x16x32_bf16(false, a, false, b1,
                                                       (short)0, acc1, false, false);
        acc2 = __builtin_amdgcn_wmma_f32_16x16x32_bf16(false, a, false, b2,
                                                       (short)0, acc2, false, false);
        acc3 = __builtin_amdgcn_wmma_f32_16x16x32_bf16(false, a, false, b3,
                                                       (short)0, acc3, false, false);
    }

    // C/D layout: VGPR r -> row (mt*16 + kh*8 + r), col = cb + tile*16 + l15
    #pragma unroll
    for (int r = 0; r < 8; ++r) {
        int row = mt * 16 + kh * 8 + r;
        int c0 = cb + l15;
        float bv0 = bias ? bias[c0]      : 0.f;
        float bv1 = bias ? bias[c0 + 16] : 0.f;
        float bv2 = bias ? bias[c0 + 32] : 0.f;
        float bv3 = bias ? bias[c0 + 48] : 0.f;
        float* cp = C + (size_t)row * N + c0;
        cp[0]  = acc0[r] + bv0;
        cp[16] = acc1[r] + bv1;
        cp[32] = acc2[r] + bv2;
        cp[48] = acc3[r] + bv3;
    }
}

// ---------------------------------------------------------------------------
// BatchNorm statistics over the (4096, 256) view of q.
// One block per channel; outputs fused scale/shift.
// ---------------------------------------------------------------------------
__global__ void k_bn_stats(const float* __restrict__ q,
                           float* __restrict__ scale, float* __restrict__ shift,
                           const float* __restrict__ gamma,
                           const float* __restrict__ beta) {
    const int c = blockIdx.x;
    const int t = threadIdx.x;
    float s = 0.f, ss = 0.f;
    for (int r = t; r < NROWS; r += 256) {
        int bs = r >> 3, h = r & 7;
        float v = q[(size_t)bs * HDQ + h * DQ + c];
        s += v; ss += v * v;
    }
    __shared__ float rs[256], rss[256];
    rs[t] = s; rss[t] = ss; __syncthreads();
    for (int o = 128; o > 0; o >>= 1) {
        if (t < o) { rs[t] += rs[t + o]; rss[t] += rss[t + o]; }
        __syncthreads();
    }
    if (t == 0) {
        float mean = rs[0] * (1.0f / NROWS);
        float var  = rss[0] * (1.0f / NROWS) - mean * mean;
        float inv  = rsqrtf(var + 1e-5f);
        float sc   = inv * gamma[c];
        scale[c] = sc;
        shift[c] = beta[c] - mean * sc;
    }
}

// ---------------------------------------------------------------------------
// Apply BN, split into halves, L2-normalize, emit bf16 query halves.
// One block (128 threads) per score row.
// ---------------------------------------------------------------------------
__global__ void k_row_norm(const float* __restrict__ q,
                           const float* __restrict__ scale,
                           const float* __restrict__ shift,
                           unsigned short* __restrict__ q0n,
                           unsigned short* __restrict__ q1n) {
    const int r = blockIdx.x;
    const int t = threadIdx.x;
    const int bs = r >> 3, h = r & 7;
    const float* base = q + (size_t)bs * HDQ + h * DQ;
    float y0 = base[t]       * scale[t]       + shift[t];
    float y1 = base[t + 128] * scale[t + 128] + shift[t + 128];
    __shared__ float s0[128], s1[128];
    s0[t] = y0 * y0; s1[t] = y1 * y1; __syncthreads();
    for (int o = 64; o > 0; o >>= 1) {
        if (t < o) { s0[t] += s0[t + o]; s1[t] += s1[t + o]; }
        __syncthreads();
    }
    __shared__ float inv0, inv1;
    if (t == 0) {
        inv0 = 1.0f / fmaxf(sqrtf(s0[0]), 1e-12f);
        inv1 = 1.0f / fmaxf(sqrtf(s1[0]), 1e-12f);
    }
    __syncthreads();
    q0n[(size_t)r * 128 + t] = f2bf(y0 * inv0);
    q1n[(size_t)r * 128 + t] = f2bf(y1 * inv1);
}

// ---------------------------------------------------------------------------
// L2-normalize the sub-key tables, emit bf16. Rows 0..255 -> k0, 256..511 -> k1.
// ---------------------------------------------------------------------------
__global__ void k_key_norm(const float* __restrict__ k0,
                           const float* __restrict__ k1,
                           unsigned short* __restrict__ k0n,
                           unsigned short* __restrict__ k1n) {
    const int row = blockIdx.x;
    const int t = threadIdx.x;
    const float* src = (row < PK) ? (k0 + (size_t)row * 128)
                                  : (k1 + (size_t)(row - PK) * 128);
    unsigned short* dst = (row < PK) ? (k0n + (size_t)row * 128)
                                     : (k1n + (size_t)(row - PK) * 128);
    float v = src[t];
    __shared__ float s[128];
    s[t] = v * v; __syncthreads();
    for (int o = 64; o > 0; o >>= 1) {
        if (t < o) s[t] += s[t + o];
        __syncthreads();
    }
    __shared__ float inv;
    if (t == 0) inv = 1.0f / fmaxf(sqrtf(s[0]), 1e-12f);
    __syncthreads();
    dst[t] = f2bf(v * inv);
}

// ---------------------------------------------------------------------------
// Per-row: top16(s0), top16(s1), 16x16 cartesian combine, top16 again,
// softmax -> weights + expert indices. One 256-thread block per row.
// Iterative argmax; on ties the lower tree slot wins, not always the lowest index.
// ---------------------------------------------------------------------------
__global__ void k_topk_combine(const float* __restrict__ s0,
                               const float* __restrict__ s1,
                               float* __restrict__ wout,
                               int* __restrict__ fiout) {
    const int r = blockIdx.x;
    const int t = threadIdx.x;
    __shared__ float arr[256];
    __shared__ float rv[256];
    __shared__ int   ri[256];
    __shared__ float ts0[TOPK], ts1[TOPK];
    __shared__ int   ti0[TOPK], ti1[TOPK];
    __shared__ float fs[TOPK];
    __shared__ int   fidx[TOPK];
    const float NEGINF = -__builtin_inff();

    // ---- top16 of s0 row ----
    arr[t] = s0[(size_t)r * PK + t]; __syncthreads();
    for (int it = 0; it < TOPK; ++it) {
        rv[t] = arr[t]; ri[t] = t; __syncthreads();
        for (int o = 128; o > 0; o >>= 1) {
            if (t < o && rv[t + o] > rv[t]) { rv[t] = rv[t + o]; ri[t] = ri[t + o]; }
            __syncthreads();
        }
        if (t == 0) { ts0[it] = rv[0]; ti0[it] = ri[0]; arr[ri[0]] = NEGINF; }
        __syncthreads();
    }
    // ---- top16 of s1 row ----
    arr[t] = s1[(size_t)r * PK + t]; __syncthreads();
    for (int it = 0; it < TOPK; ++it) {
        rv[t] = arr[t]; ri[t] = t; __syncthreads();
        for (int o = 128; o > 0; o >>= 1) {
            if (t < o && rv[t + o] > rv[t]) { rv[t] = rv[t + o]; ri[t] = ri[t + o]; }
            __syncthreads();
        }
        if (t == 0) { ts1[it] = rv[0]; ti1[it] = ri[0]; arr[ri[0]] = NEGINF; }
        __syncthreads();
    }
    // ---- 256 cartesian sums, top16 of those ----
    {
        int i = t >> 4, j = t & 15;
        arr[t] = ts0[i] + ts1[j];
    }
    __syncthreads();
    for (int it = 0; it < TOPK; ++it) {
        rv[t] = arr[t]; ri[t] = t; __syncthreads();
        for (int o = 128; o > 0; o >>= 1) {
            if (t < o && rv[t + o] > rv[t]) { rv[t] = rv[t + o]; ri[t] = ri[t + o]; }
            __syncthreads();
        }
        if (t == 0) {
            int p = ri[0];
            fs[it]   = rv[0];
            fidx[it] = ti0[p >> 4] * PK + ti1[p & 15];
            arr[p] = NEGINF;
        }
        __syncthreads();
    }
    // ---- softmax + emit ----
    if (t == 0) {
        float m = fs[0];
        #pragma unroll
        for (int k = 1; k < TOPK; ++k) m = fmaxf(m, fs[k]);
        float e[TOPK]; float sum = 0.f;
        #pragma unroll
        for (int k = 0; k < TOPK; ++k) { e[k] = expf(fs[k] - m); sum += e[k]; }
        float invs = 1.0f / sum;
        #pragma unroll
        for (int k = 0; k < TOPK; ++k) {
            wout [(size_t)r * TOPK + k] = e[k] * invs;
            fiout[(size_t)r * TOPK + k] = fidx[k];
        }
    }
}

// ---------------------------------------------------------------------------
// Expert gather + apply. One 256-thread (8-wave) block per token (b,s).
// 128 selected experts per token; each wave handles 16 of them.
// Pure-bandwidth phase (~512 MB of gathers): each lane owns 4 consecutive
// floats (d = (i*32+lane)*4) so every row read is 8 coalesced
// global_load_b128 per lane, with non-temporal hints (single-use data,
// keep L2 for the reused tensors). Waves merge via LDS ds_add_f32.
// ---------------------------------------------------------------------------
__global__ void k_expert_moe(const float* __restrict__ hidden,
                             const float* __restrict__ down,
                             const float* __restrict__ up,
                             const float* __restrict__ w,
                             const int* __restrict__ fi,
                             float* __restrict__ out) {
    const int bs = blockIdx.x;
    const int t = threadIdx.x;
    const int lane = t & 31;
    const int wv = t >> 5;

    __shared__ float hs[DIN];
    __shared__ float acc[DOUT];
    __shared__ int   eidx[NH * TOPK];
    __shared__ float ewt [NH * TOPK];

    // stage hidden vector + zero accumulator (one f4v per thread)
    {
        f4v* hs4  = (f4v*)hs;
        f4v* acc4 = (f4v*)acc;
        const f4v* hp4 = (const f4v*)(hidden + (size_t)bs * DIN);
        hs4[t]  = hp4[t];
        acc4[t] = (f4v){0.f, 0.f, 0.f, 0.f};
    }
    if (t < NH * TOPK) {
        int h = t >> 4, k = t & 15;
        int r = bs * NH + h;
        eidx[t] = fi[(size_t)r * TOPK + k];
        ewt[t]  = w [(size_t)r * TOPK + k];
    }
    __syncthreads();

    f4v accr[8];
    #pragma unroll
    for (int i = 0; i < 8; ++i) accr[i] = (f4v){0.f, 0.f, 0.f, 0.f};

    const f4v* hs4 = (const f4v*)hs;

    for (int j = wv; j < NH * TOPK; j += 8) {
        int e = eidx[j];
        const f4v* dp4  = (const f4v*)(down + (size_t)e * DIN);
        const f4v* upp4 = (const f4v*)(up   + (size_t)e * DOUT);
        if (j + 8 < NH * TOPK) {
            int en = eidx[j + 8];
            __builtin_prefetch(down + (size_t)en * DIN  + lane * 32, 0, 1);
            __builtin_prefetch(up   + (size_t)en * DOUT + lane * 32, 0, 1);
        }
        // dot(down[e], hidden) : lane owns 8 f4v chunks at (i*32+lane)
        float d = 0.f;
        #pragma unroll
        for (int i = 0; i < 8; ++i) {
            f4v dv = __builtin_nontemporal_load(dp4 + i * 32 + lane);
            f4v hv = hs4[i * 32 + lane];
            d += dv.x * hv.x + dv.y * hv.y + dv.z * hv.z + dv.w * hv.w;
        }
        #pragma unroll
        for (int o = 16; o > 0; o >>= 1) d += __shfl_xor(d, o, 32);
        float g = 0.5f * d * (1.0f + erff(d * 0.70710678118654752f));  // exact GELU
        g *= ewt[j];
        // weighted up-row accumulate
        #pragma unroll
        for (int i = 0; i < 8; ++i) {
            f4v uv = __builtin_nontemporal_load(upp4 + i * 32 + lane);
            accr[i] += uv * g;
        }
    }

    // merge 8 waves through LDS float atomics
    #pragma unroll
    for (int i = 0; i < 8; ++i) {
        int base = (i * 32 + lane) * 4;
        atomicAdd(&acc[base + 0], accr[i].x);
        atomicAdd(&acc[base + 1], accr[i].y);
        atomicAdd(&acc[base + 2], accr[i].z);
        atomicAdd(&acc[base + 3], accr[i].w);
    }
    __syncthreads();
    {
        f4v* out4 = (f4v*)(out + (size_t)bs * DOUT);
        out4[t] = ((const f4v*)acc)[t];
    }
}

// ---------------------------------------------------------------------------
// Host launch
// ---------------------------------------------------------------------------
extern "C" void kernel_launch(void* const* d_in, const int* in_sizes, int n_in,
                              void* d_out, int out_size, void* d_ws, size_t ws_size,
                              hipStream_t stream) {
    const float* hidden = (const float*)d_in[0];   // [2,256,1024]
    const float* Wq     = (const float*)d_in[1];   // [2048,1024]
    const float* bq     = (const float*)d_in[2];   // [2048]
    const float* gamma  = (const float*)d_in[3];   // [256]
    const float* beta   = (const float*)d_in[4];   // [256]
    const float* sk0    = (const float*)d_in[5];   // [256,128]
    const float* sk1    = (const float*)d_in[6];   // [256,128]
    const float* edown  = (const float*)d_in[7];   // [65536,1024]
    const float* eup    = (const float*)d_in[8];   // [65536,1024]
    float* out = (float*)d_out;

    char* ws = (char*)d_ws;
    size_t off = 0;
    auto alloc = [&](size_t bytes) -> char* {
        char* p = ws + off;
        off = (off + bytes + 255) & ~(size_t)255;
        return p;
    };
    unsigned short* hbf  = (unsigned short*)alloc((size_t)NTOK * DIN * 2);
    unsigned short* wqbf = (unsigned short*)alloc((size_t)HDQ * DIN * 2);
    float*          qbuf = (float*)alloc((size_t)NTOK * HDQ * 4);
    float*          scal = (float*)alloc(DQ * 4);
    float*          shft = (float*)alloc(DQ * 4);
    unsigned short* q0n  = (unsigned short*)alloc((size_t)NROWS * 128 * 2);
    unsigned short* q1n  = (unsigned short*)alloc((size_t)NROWS * 128 * 2);
    unsigned short* k0n  = (unsigned short*)alloc((size_t)PK * 128 * 2);
    unsigned short* k1n  = (unsigned short*)alloc((size_t)PK * 128 * 2);
    float*          s0b  = (float*)alloc((size_t)NROWS * PK * 4);
    float*          s1b  = (float*)alloc((size_t)NROWS * PK * 4);
    float*          wbuf = (float*)alloc((size_t)NROWS * TOPK * 4);
    int*            fib  = (int*)alloc((size_t)NROWS * TOPK * 4);

    // 1) bf16 conversion of GEMM operands
    {
        int n = NTOK * DIN;
        k_cvt_bf16<<<(n + 255) / 256, 256, 0, stream>>>(hidden, hbf, n);
        n = HDQ * DIN;
        k_cvt_bf16<<<(n + 255) / 256, 256, 0, stream>>>(Wq, wqbf, n);
    }

    // 2) q = hidden @ Wq^T + bq      (WMMA bf16, f32 accumulate)
    {
        int waves = (NTOK / 16) * (HDQ / 64);          // 32*32 = 1024
        k_wmma_gemm_bf16<<<(waves + 7) / 8, 256, 0, stream>>>(
            hbf, wqbf, qbuf, bq, NTOK, HDQ, DIN);
    }

    // 3) BatchNorm stats + apply/L2norm
    k_bn_stats<<<DQ, 256, 0, stream>>>(qbuf, scal, shft, gamma, beta);
    k_row_norm<<<NROWS, 128, 0, stream>>>(qbuf, scal, shft, q0n, q1n);
    k_key_norm<<<2 * PK, 128, 0, stream>>>(sk0, sk1, k0n, k1n);

    // 4) score GEMMs (WMMA bf16)
    {
        int waves = (NROWS / 16) * (PK / 64);          // 256*4 = 1024
        k_wmma_gemm_bf16<<<(waves + 7) / 8, 256, 0, stream>>>(
            q0n, k0n, s0b, nullptr, NROWS, PK, 128);
        k_wmma_gemm_bf16<<<(waves + 7) / 8, 256, 0, stream>>>(
            q1n, k1n, s1b, nullptr, NROWS, PK, 128);
    }

    // 5) top-k / combine / softmax
    k_topk_combine<<<NROWS, 256, 0, stream>>>(s0b, s1b, wbuf, fib);

    // 6) expert gather + apply (bandwidth-bound, ~512 MB of gathers)
    k_expert_moe<<<NTOK, 256, 0, stream>>>(hidden, edown, eup, wbuf, fib, out);

    (void)in_sizes; (void)n_in; (void)out_size; (void)ws_size;
}
